// EncoderLayer_30030411333992
// MI455X (gfx1250) — matrix operand order repaired and checked by reading, not compiled, not run on hardware
//
#include <hip/hip_runtime.h>
#include <hip/hip_bf16.h>
#include <math.h>
#include <stdint.h>

typedef __bf16 bf16;
typedef __attribute__((ext_vector_type(8)))  __bf16 v8bf;
typedef __attribute__((ext_vector_type(16))) __bf16 v16bf;
typedef __attribute__((ext_vector_type(8)))  float  v8f;

#define HID   1024
#define SEQ   2048
#define BATCH 4
#define HEADS 16
#define HDIM  64
#define PFF   4096
#define TOK   (BATCH * SEQ)   // 8192 tokens

// ---------------------------------------------------------------------------
// CDNA5 primitives
// ---------------------------------------------------------------------------
__device__ __forceinline__ v8f wmma_bf16(v16bf a, v16bf b, v8f c) {
  return __builtin_amdgcn_wmma_f32_16x16x32_bf16(false, a, false, b,
                                                 (short)0, c, false, false);
}

// Async copy 16B global -> LDS (per lane), tracked by ASYNCcnt.
// Generic pointers to LDS carry the byte offset in their low 32 bits
__device__ __forceinline__ void async_copy_b128(const bf16* gptr, void* lptr) {
  unsigned lds = (unsigned)(uintptr_t)lptr;
  asm volatile("global_load_async_to_lds_b128 %0, %1, off"
               :: "v"(lds), "v"(gptr) : "memory");
}
__device__ __forceinline__ void wait_async_le4() {
  asm volatile("s_wait_asynccnt 0x4" ::: "memory");
}
__device__ __forceinline__ void wait_async_le1() {
  asm volatile("s_wait_asynccnt 0x1" ::: "memory");
}
__device__ __forceinline__ void wait_async_le0() {
  asm volatile("s_wait_asynccnt 0x0" ::: "memory");
}

__device__ __forceinline__ v16bf pack16(v8bf lo, v8bf hi) {
  v16bf r;
  #pragma unroll
  for (int e = 0; e < 8; e++) { r[e] = lo[e]; r[8 + e] = hi[e]; }
  return r;
}

// A-operand fragment: lane m = lane%16, h = lane/16.
// K elements: [8h .. 8h+7] then [16+8h .. 16+8h+7]  (ISA 16-bit A 16x32 table)
__device__ __forceinline__ v16bf load_afrag(const bf16* rowp, int h) {
  v8bf lo = *(const v8bf*)(rowp + 8 * h);
  v8bf hi = *(const v8bf*)(rowp + 16 + 8 * h);
  return pack16(lo, hi);
}

// B-operand fragment (row of K-major data): lane n = lane%16.
// K elements: [8h .. 8h+7] then [16+8h .. 16+8h+7]  (same order as A fragment)
__device__ __forceinline__ v16bf load_bfrag(const bf16* rowp, int h) {
  v8bf lo = *(const v8bf*)(rowp + 8 * h);
  v8bf hi = *(const v8bf*)(rowp + 16 + 8 * h);
  return pack16(lo, hi);
}

// ---------------------------------------------------------------------------
// f32 -> bf16 elementwise convert
// ---------------------------------------------------------------------------
__global__ void k_cvt_bf16(const float* __restrict__ in, bf16* __restrict__ out,
                           int n) {
  int i = blockIdx.x * blockDim.x + threadIdx.x;
  if (i < n) out[i] = (bf16)in[i];
}

// int mask -> additive f32 bias (0 or -1e10)
__global__ void k_maskbias(const int* __restrict__ m, float* __restrict__ out,
                           int n) {
  int i = blockIdx.x * blockDim.x + threadIdx.x;
  if (i < n) out[i] = m[i] ? 0.0f : -1e10f;
}

// ---------------------------------------------------------------------------
// W (KxN f32, row-major) -> Wt (NxK bf16)  via LDS-tiled transpose
// ---------------------------------------------------------------------------
__global__ __launch_bounds__(256) void k_transpose_cvt(
    const float* __restrict__ in, bf16* __restrict__ out, int K, int N) {
  __shared__ float t[32][33];
  const int k0 = blockIdx.x * 32, n0 = blockIdx.y * 32;
  const int tx = threadIdx.x, ty = threadIdx.y;   // 32 x 8
  #pragma unroll
  for (int r = ty; r < 32; r += 8)
    t[r][tx] = in[(size_t)(k0 + r) * N + (n0 + tx)];
  __syncthreads();
  #pragma unroll
  for (int r = ty; r < 32; r += 8)
    out[(size_t)(n0 + r) * K + (k0 + tx)] = (bf16)t[tx][r];
}

// ---------------------------------------------------------------------------
// GEMM: C[M,N] = A[M,K](bf16) * Bt[N,K](bf16)^T + bias
// 128x128 block tile, K-step 32, 8 waves (each 32x64 = 2x4 WMMA tiles).
// Double-buffered LDS fed by global_load_async_to_lds_b128.
// EPI: 0 = bias -> bf16 | 1 = bias + residual(f32) -> f32
//      2 = swish(bias) -> bf16 | 3 = (bias) * 1/sqrt(HDIM) -> bf16 (Q proj)
// ---------------------------------------------------------------------------
template <int EPI>
__global__ __launch_bounds__(256) void k_gemm(
    const bf16* __restrict__ A, const bf16* __restrict__ Bt,
    const float* __restrict__ bias, const float* __restrict__ res,
    float* __restrict__ outF, bf16* __restrict__ outB,
    int M, int N, int K) {
  __shared__ bf16 sA[2][128][40];   // 80B row stride: conflict-free b128 reads
  __shared__ bf16 sB[2][128][40];

  const int tid  = threadIdx.x;
  const int m0   = blockIdx.y * 128, n0 = blockIdx.x * 128;
  const int wave = tid >> 5, lane = tid & 31;
  const int lm   = lane & 15, h = lane >> 4;
  const int wm   = wave & 3, wn = wave >> 2;    // 4(M) x 2(N) wave grid

  // this thread's two 16B chunks of each tile
  const int r0 = tid >> 2,          kc0 = (tid & 3) * 8;
  const int r1 = (tid + 256) >> 2,  kc1 = ((tid + 256) & 3) * 8;

  v8f acc[2][4] = {};
  const int nk = K >> 5;

  // prologue: stream tile 0 into buffer 0 (4 async ops / thread)
  async_copy_b128(&A [(size_t)(m0 + r0) * K + kc0], &sA[0][r0][kc0]);
  async_copy_b128(&A [(size_t)(m0 + r1) * K + kc1], &sA[0][r1][kc1]);
  async_copy_b128(&Bt[(size_t)(n0 + r0) * K + kc0], &sB[0][r0][kc0]);
  async_copy_b128(&Bt[(size_t)(n0 + r1) * K + kc1], &sB[0][r1][kc1]);

  for (int it = 0; it < nk; it++) {
    const int cur = it & 1;
    if (it + 1 < nk) {                    // prefetch next tile into other buf
      const int nb = (it + 1) & 1;
      const int k0 = (it + 1) << 5;
      async_copy_b128(&A [(size_t)(m0 + r0) * K + k0 + kc0], &sA[nb][r0][kc0]);
      async_copy_b128(&A [(size_t)(m0 + r1) * K + k0 + kc1], &sA[nb][r1][kc1]);
      async_copy_b128(&Bt[(size_t)(n0 + r0) * K + k0 + kc0], &sB[nb][r0][kc0]);
      async_copy_b128(&Bt[(size_t)(n0 + r1) * K + k0 + kc1], &sB[nb][r1][kc1]);
      wait_async_le4();                   // current tile's 4 ops complete
    } else {
      wait_async_le0();
    }
    __syncthreads();                      // all waves' tile data visible

    v16bf af[2];
    #pragma unroll
    for (int am = 0; am < 2; am++)
      af[am] = load_afrag(&sA[cur][wm * 32 + am * 16 + lm][0], h);

    #pragma unroll
    for (int bn = 0; bn < 4; bn++) {
      v16bf bfr = load_bfrag(&sB[cur][wn * 64 + bn * 16 + lm][0], h);
      acc[0][bn] = wmma_bf16(af[0], bfr, acc[0][bn]);
      acc[1][bn] = wmma_bf16(af[1], bfr, acc[1][bn]);
    }
    __syncthreads();                      // free 'cur' for overwrite
  }

  // epilogue (C layout: VGPR r -> row r + 8h, col = lane%16)
  #pragma unroll
  for (int am = 0; am < 2; am++) {
    #pragma unroll
    for (int bn = 0; bn < 4; bn++) {
      const int gn = n0 + wn * 64 + bn * 16 + lm;
      const float bv = bias[gn];
      #pragma unroll
      for (int r = 0; r < 8; r++) {
        const int gm = m0 + wm * 32 + am * 16 + r + 8 * h;
        const size_t off = (size_t)gm * N + gn;
        float v = acc[am][bn][r] + bv;
        if (EPI == 0) {
          outB[off] = (bf16)v;
        } else if (EPI == 1) {
          outF[off] = v + res[off];
        } else if (EPI == 2) {
          outB[off] = (bf16)(v / (1.0f + __expf(-v)));   // swish
        } else {
          outB[off] = (bf16)(v * 0.125f);                // pre-scaled Q
        }
      }
    }
  }
}

// ---------------------------------------------------------------------------
// Fused flash attention, online softmax, double-buffered K/V tiles.
// Q is pre-scaled by 1/sqrt(HDIM); mask is an additive f32 bias.
// Block = (128 q-rows, head, batch); 8 waves, 16 q-rows each.
// Per 32-key tile: 4 score WMMAs + 4 P*V WMMAs.
// ---------------------------------------------------------------------------
__global__ __launch_bounds__(256) void k_flash(
    const bf16* __restrict__ Q, const bf16* __restrict__ Km,
    const bf16* __restrict__ Vm, const float* __restrict__ maskAdd,
    bf16* __restrict__ O) {
  __shared__ bf16 sK[2][32][72];     // [buf][key][dim]
  __shared__ bf16 sV[2][64][40];     // [buf][dim][key] (transposed)
  __shared__ bf16 sP[8][16][40];     // per-wave P staging [m][k]

  const int tid  = threadIdx.x;
  const int wave = tid >> 5, lane = tid & 31;
  const int lm   = lane & 15, h = lane >> 4;
  const int b    = blockIdx.z, head = blockIdx.y;
  const int q0   = blockIdx.x * 128;
  const int NT   = SEQ / 32;

  // Q fragments in registers (head_dim 64 -> two K=32 chunks)
  v16bf qf[2];
  {
    const bf16* qrow =
        Q + (size_t)(b * SEQ + q0 + wave * 16 + lm) * HID + head * HDIM;
    qf[0] = load_afrag(qrow, h);
    qf[1] = load_afrag(qrow + 32, h);
  }

  v8f o[4] = {};
  float mstat[8], lstat[8];
  #pragma unroll
  for (int r = 0; r < 8; r++) { mstat[r] = -1e30f; lstat[r] = 0.0f; }

  const int ck = tid & 7, key = tid >> 3;   // 16B chunk mapping
  const size_t kvBase = (size_t)(b * SEQ + key) * HID + head * HDIM + ck * 8;

  // prologue: async K tile 0 -> sK[0]; V tile 0 -> registers
  async_copy_b128(Km + kvBase, &sK[0][key][ck * 8]);
  v8bf vv = *(const v8bf*)(Vm + kvBase);

  for (int kt = 0; kt < NT; kt++) {
    const int cur = kt & 1;
    // scatter this tile's V registers into sV[cur] (transpose)
    #pragma unroll
    for (int e = 0; e < 8; e++) sV[cur][ck * 8 + e][key] = vv[e];

    if (kt + 1 < NT) {                     // prefetch next K/V tile
      const size_t noff = kvBase + (size_t)(kt + 1) * 32 * HID;
      async_copy_b128(Km + noff, &sK[(kt + 1) & 1][key][ck * 8]);
      vv = *(const v8bf*)(Vm + noff);
      wait_async_le1();                    // tile kt's K resident
    } else {
      wait_async_le0();
    }
    __syncthreads();

    // scores S = Q * K^T  (two 16-key subtiles, 2 head-dim chunks each)
    v8f s[2];
    #pragma unroll
    for (int nt = 0; nt < 2; nt++) {
      v8f a = {};
      const bf16* krow = &sK[cur][nt * 16 + lm][0];
      a = wmma_bf16(qf[0], load_bfrag(krow, h), a);
      a = wmma_bf16(qf[1], load_bfrag(krow + 32, h), a);
      s[nt] = a;
    }

    // additive mask (0 / -1e10); scale already folded into Q
    const float ma0 = maskAdd[b * SEQ + kt * 32 + lm];
    const float ma1 = maskAdd[b * SEQ + kt * 32 + 16 + lm];
    #pragma unroll
    for (int r = 0; r < 8; r++) { s[0][r] += ma0; s[1][r] += ma1; }

    // online softmax (row = r + 8h; reduce over the 16 lanes of this half)
    #pragma unroll
    for (int r = 0; r < 8; r++) {
      float v = fmaxf(s[0][r], s[1][r]);
      #pragma unroll
      for (int xm = 8; xm >= 1; xm >>= 1) v = fmaxf(v, __shfl_xor(v, xm, 32));
      const float mn = fmaxf(mstat[r], v);
      const float alpha = __expf(mstat[r] - mn);
      mstat[r] = mn;
      const float p0 = __expf(s[0][r] - mn);
      const float p1 = __expf(s[1][r] - mn);
      s[0][r] = p0; s[1][r] = p1;
      float rs = p0 + p1;
      #pragma unroll
      for (int xm = 8; xm >= 1; xm >>= 1) rs += __shfl_xor(rs, xm, 32);
      lstat[r] = lstat[r] * alpha + rs;
      o[0][r] *= alpha; o[1][r] *= alpha; o[2][r] *= alpha; o[3][r] *= alpha;
    }

    // stage P (bf16) through per-wave LDS, reload as A fragment
    #pragma unroll
    for (int nt = 0; nt < 2; nt++)
      #pragma unroll
      for (int r = 0; r < 8; r++)
        sP[wave][r + 8 * h][nt * 16 + lm] = (bf16)s[nt][r];
    v16bf pf = load_afrag(&sP[wave][lm][0], h);   // same-wave DS, in-order

    // O += P * V
    #pragma unroll
    for (int nf = 0; nf < 4; nf++)
      o[nf] = wmma_bf16(pf, load_bfrag(&sV[cur][nf * 16 + lm][0], h), o[nf]);

    __syncthreads();                       // free 'cur' buffers
  }

  // epilogue: O /= l, write bf16 into (B,S,HID) layout
  #pragma unroll
  for (int nf = 0; nf < 4; nf++) {
    #pragma unroll
    for (int r = 0; r < 8; r++) {
      const size_t off = (size_t)(b * SEQ + q0 + wave * 16 + r + 8 * h) * HID +
                         head * HDIM + nf * 16 + lm;
      O[off] = (bf16)(o[nf][r] / lstat[r]);
    }
  }
}

// ---------------------------------------------------------------------------
// LayerNorm over HID=1024; one block (256 thr) per row; f32 + optional bf16 out
// ---------------------------------------------------------------------------
__global__ __launch_bounds__(256) void k_layernorm(
    const float* __restrict__ x, const float* __restrict__ g,
    const float* __restrict__ bb, float* __restrict__ outF,
    bf16* __restrict__ outB) {
  __shared__ float red[8];
  const int row = blockIdx.x, tid = threadIdx.x;
  const float* xr = x + (size_t)row * HID;

  float v[4];
  float s = 0.0f;
  #pragma unroll
  for (int i = 0; i < 4; i++) { v[i] = xr[tid + i * 256]; s += v[i]; }
  #pragma unroll
  for (int xm = 16; xm >= 1; xm >>= 1) s += __shfl_xor(s, xm, 32);
  if ((tid & 31) == 0) red[tid >> 5] = s;
  __syncthreads();
  s = 0.0f;
  #pragma unroll
  for (int i = 0; i < 8; i++) s += red[i];
  const float mu = s * (1.0f / HID);
  __syncthreads();

  float t = 0.0f;
  #pragma unroll
  for (int i = 0; i < 4; i++) { float d = v[i] - mu; t += d * d; }
  #pragma unroll
  for (int xm = 16; xm >= 1; xm >>= 1) t += __shfl_xor(t, xm, 32);
  if ((tid & 31) == 0) red[tid >> 5] = t;
  __syncthreads();
  t = 0.0f;
  #pragma unroll
  for (int i = 0; i < 8; i++) t += red[i];
  const float inv = rsqrtf(t * (1.0f / HID) + 1e-5f);

  #pragma unroll
  for (int i = 0; i < 4; i++) {
    const int col = tid + i * 256;
    const float y = (v[i] - mu) * inv * g[col] + bb[col];
    outF[(size_t)row * HID + col] = y;
    if (outB) outB[(size_t)row * HID + col] = (bf16)y;
  }
}

// ---------------------------------------------------------------------------
// Host orchestration
// ---------------------------------------------------------------------------
extern "C" void kernel_launch(void* const* d_in, const int* in_sizes, int n_in,
                              void* d_out, int out_size, void* d_ws,
                              size_t ws_size, hipStream_t stream) {
  (void)in_sizes; (void)n_in; (void)out_size; (void)ws_size;

  const float* src  = (const float*)d_in[0];
  const int*   mask = (const int*)d_in[1];
  const float* Wq = (const float*)d_in[2];  const float* bq = (const float*)d_in[3];
  const float* Wk = (const float*)d_in[4];  const float* bk = (const float*)d_in[5];
  const float* Wv = (const float*)d_in[6];  const float* bv = (const float*)d_in[7];
  const float* Wo = (const float*)d_in[8];  const float* bo = (const float*)d_in[9];
  const float* W1 = (const float*)d_in[10]; const float* b1 = (const float*)d_in[11];
  const float* W2 = (const float*)d_in[12]; const float* b2 = (const float*)d_in[13];
  const float* ln1g = (const float*)d_in[14]; const float* ln1b = (const float*)d_in[15];
  const float* ln2g = (const float*)d_in[16]; const float* ln2b = (const float*)d_in[17];
  float* out = (float*)d_out;

  char* w = (char*)d_ws;
  const size_t MB = 1024ull * 1024ull;
  bf16*  WqT    = (bf16*)(w + 0 * MB);
  bf16*  WkT    = (bf16*)(w + 2 * MB);
  bf16*  WvT    = (bf16*)(w + 4 * MB);
  bf16*  WoT    = (bf16*)(w + 6 * MB);
  bf16*  W1T    = (bf16*)(w + 8 * MB);     // 8 MB
  bf16*  W2T    = (bf16*)(w + 16 * MB);    // 8 MB
  bf16*  srcB   = (bf16*)(w + 24 * MB);    // 16 MB
  bf16*  Qb     = (bf16*)(w + 40 * MB);    // 16 MB
  bf16*  Kb     = (bf16*)(w + 56 * MB);    // 16 MB
  bf16*  Vb     = (bf16*)(w + 72 * MB);    // 16 MB
  bf16*  attnB  = (bf16*)(w + 88 * MB);    // 16 MB
  float* resid1 = (float*)(w + 104 * MB);  // 32 MB
  float* ln1F   = (float*)(w + 136 * MB);  // 32 MB
  bf16*  ln1B   = (bf16*)(w + 168 * MB);   // 16 MB
  float* maskF  = (float*)(w + 184 * MB);  // 32 KB  -> peak ~184 MB
  bf16*  hB     = srcB;    // reuse [24,88): src/Q/K/V dead after attention
  float* resid2 = resid1;  // resid1 dead after LN1

  // 1) precision conversion, mask lowering, weight transposes
  const int nsrc = TOK * HID;
  k_cvt_bf16<<<(nsrc + 255) / 256, 256, 0, stream>>>(src, srcB, nsrc);
  k_maskbias<<<(BATCH * SEQ + 255) / 256, 256, 0, stream>>>(mask, maskF,
                                                            BATCH * SEQ);
  dim3 tb(32, 8);
  k_transpose_cvt<<<dim3(32, 32), tb, 0, stream>>>(Wq, WqT, HID, HID);
  k_transpose_cvt<<<dim3(32, 32), tb, 0, stream>>>(Wk, WkT, HID, HID);
  k_transpose_cvt<<<dim3(32, 32), tb, 0, stream>>>(Wv, WvT, HID, HID);
  k_transpose_cvt<<<dim3(32, 32), tb, 0, stream>>>(Wo, WoT, HID, HID);
  k_transpose_cvt<<<dim3(32, 128), tb, 0, stream>>>(W1, W1T, HID, PFF);
  k_transpose_cvt<<<dim3(128, 32), tb, 0, stream>>>(W2, W2T, PFF, HID);

  // 2) QKV projections (Q pre-scaled by 1/sqrt(HDIM))
  dim3 gqkv(HID / 128, TOK / 128);
  k_gemm<3><<<gqkv, 256, 0, stream>>>(srcB, WqT, bq, nullptr, nullptr, Qb,
                                      TOK, HID, HID);
  k_gemm<0><<<gqkv, 256, 0, stream>>>(srcB, WkT, bk, nullptr, nullptr, Kb,
                                      TOK, HID, HID);
  k_gemm<0><<<gqkv, 256, 0, stream>>>(srcB, WvT, bv, nullptr, nullptr, Vb,
                                      TOK, HID, HID);

  // 3) fused attention
  k_flash<<<dim3(SEQ / 128, HEADS, BATCH), 256, 0, stream>>>(Qb, Kb, Vb, maskF,
                                                             attnB);

  // 4) output projection + residual (f32 src)
  k_gemm<1><<<gqkv, 256, 0, stream>>>(attnB, WoT, bo, src, resid1, nullptr,
                                      TOK, HID, HID);

  // 5) LN1 -> f32 (residual source) + bf16 (next GEMM operand)
  k_layernorm<<<TOK, 256, 0, stream>>>(resid1, ln1g, ln1b, ln1F, ln1B);

  // 6) FFN up + swish
  k_gemm<2><<<dim3(PFF / 128, TOK / 128), 256, 0, stream>>>(
      ln1B, W1T, b1, nullptr, nullptr, hB, TOK, PFF, HID);

  // 7) FFN down + residual (f32 ln1 output)
  k_gemm<1><<<gqkv, 256, 0, stream>>>(hB, W2T, b2, ln1F, resid2, nullptr,
                                      TOK, HID, PFF);

  // 8) LN2 -> final f32 output
  k_layernorm<<<TOK, 256, 0, stream>>>(resid2, ln2g, ln2b, out, nullptr);
}
